// NodeClassificationMPNTag_35923106464068
// MI455X (gfx1250) — matrix operand order repaired and checked by reading, not compiled, not run on hardware
//
#include <hip/hip_runtime.h>
#include <hip/hip_bf16.h>
#include <stdint.h>

// ---------------------------------------------------------------------------
// NodeClassificationMPNTag on MI455X (gfx1250).
//
// ~147 GFLOP of GEMM vs ~1.2 GB HBM traffic -> f16 storage with
// v_wmma_f32_16x16x32_f16 (f32 accumulate). The 768-wide edge-MLP input is
// gathered straight into LDS (never materialized in HBM). This revision
// register-blocks M 4x (64 edges / block): each B fragment (weight slice)
// feeds 4 WMMAs, cutting L2 weight traffic 4x vs the previous version.
// segment_sum is global_atomic_add_f32 into an f32 buffer.
// ---------------------------------------------------------------------------

#define NN 10000     // N_NODES
#define NE 160000    // N_EDGES

typedef __attribute__((ext_vector_type(16))) _Float16 v16h;
typedef __attribute__((ext_vector_type(8)))  _Float16 v8h;
typedef __attribute__((ext_vector_type(8)))  float    v8f;

union AFrag { v16h v; v8h h[2]; };

// ---- WMMA 16x16x32 f16 tile helpers (ISA 7.12.2 layouts, wave32) ----------
// A (16x32, MxK): lane L<16 holds row M=L, K=0..7 & 16..23; lane L>=16 holds
// row M=L-16, K=8..15 & 24..31.
__device__ __forceinline__ v16h load_a_tile(const _Float16* tile, int lda) {
  const int lane = threadIdx.x & 31;
  const int r    = lane & 15;
  const int hi   = lane >> 4;
  const _Float16* p = tile + (size_t)r * lda + hi * 8;
  AFrag a;
  a.h[0] = *(const v8h*)(p);
  a.h[1] = *(const v8h*)(p + 16);
  return a.v;
}

// B (32x16, KxN), W row-major [K,128]: lane L holds col N=n0+(L&15), K=k0+8*(L>>4)+0..7 & k0+16+8*(L>>4)+0..7
__device__ __forceinline__ v16h load_b_tile(const _Float16* W, int k0, int n0) {
  const int lane = threadIdx.x & 31;
  const int col  = lane & 15;
  const int hi   = lane >> 4;
  v16h b;
#pragma unroll
  for (int s = 0; s < 8; ++s) {
    b[s]     = W[(size_t)(k0 + 8 * hi + s) * 128 + n0 + col];
    b[8 + s] = W[(size_t)(k0 + 16 + 8 * hi + s) * 128 + n0 + col];
  }
  return b;
}

__device__ __forceinline__ v8f wmma_step(v16h a, v16h b, v8f c) {
  return __builtin_amdgcn_wmma_f32_16x16x32_f16(
      /*neg_a=*/false, a, /*neg_b=*/false, b,
      /*c_mod=*/(short)0, c, /*reuse_a=*/false, /*reuse_b=*/false);
}

// C/D (16x16 f32): VGPR r -> row r + 8*(lane>=16), col = lane&15
__device__ __forceinline__ void store_c_f16(v8f c, const float* __restrict__ bias,
                                            _Float16* __restrict__ out,
                                            size_t m0, int n0, int relu) {
  const int lane = threadIdx.x & 31;
  const int col  = lane & 15;
  const int hi   = lane >> 4;
#pragma unroll
  for (int r = 0; r < 8; ++r) {
    float v = c[r] + bias[n0 + col];
    if (relu) v = v > 0.f ? v : 0.f;
    out[(m0 + (size_t)(r + 8 * hi)) * 128 + n0 + col] = (_Float16)v;
  }
}

// ---- generic tiled GEMM: out[M,128] = act(A[M,K] @ W[K,128] + b) -----------
// grid.x = M/(16*TM); 256 threads = 8 waves, wave w owns N-tile n0 = 16*w and
// TM accumulator tiles in M (B fragment reused TM times per k-step).
template <int TM>
__global__ void gemm_tiled(const _Float16* __restrict__ A,
                           const _Float16* __restrict__ W,
                           const float* __restrict__ bias,
                           _Float16* __restrict__ out,
                           int K, int relu) {
  extern __shared__ _Float16 sA[];
  const int tid = threadIdx.x;
  const size_t m0 = (size_t)blockIdx.x * (16 * TM);
  const int total = 16 * TM * K;
  for (int i = tid * 8; i < total; i += 256 * 8)
    *(v8h*)(sA + i) = *(const v8h*)(A + m0 * (size_t)K + i);
  __syncthreads();

  const int n0 = (tid >> 5) * 16;
  v8f c[TM];
#pragma unroll
  for (int mt = 0; mt < TM; ++mt) c[mt] = (v8f){};

  for (int k0 = 0; k0 < K; k0 += 32) {
    v16h b = load_b_tile(W, k0, n0);
#pragma unroll
    for (int mt = 0; mt < TM; ++mt) {
      v16h a = load_a_tile(sA + (size_t)mt * 16 * K + k0, K);
      c[mt] = wmma_step(a, b, c[mt]);
    }
  }
#pragma unroll
  for (int mt = 0; mt < TM; ++mt)
    store_c_f16(c[mt], bias, out, m0 + 16 * mt, n0, relu);
}

// ---- fused edge MLP + scatter-add aggregation ------------------------------
// 64 edges / block: gather msg_in (64x768) into LDS from 6 segments,
// H = relu(msg_in @ W1 + b1)  (64x128, LDS),
// ef_new = relu(H @ W2 + b2)  -> global f16 + atomic f32 add into aggr[dst].
// 8 waves x (4 M-tiles, 1 N-tile); each B fragment feeds 4 WMMAs.
__global__ void mp_edge_kernel(const _Float16* __restrict__ nf0,
                               const _Float16* __restrict__ nf,
                               const _Float16* __restrict__ ef0,
                               const _Float16* __restrict__ ef,
                               const int* __restrict__ edge_index,
                               const _Float16* __restrict__ W1,
                               const float* __restrict__ b1,
                               const _Float16* __restrict__ W2,
                               const float* __restrict__ b2,
                               _Float16* __restrict__ ef_out,
                               float* __restrict__ aggr) {
  extern __shared__ _Float16 smem[];            // 64*768 (msg) + 64*128 (hidden)
  _Float16* sA = smem;                          // 96 KB
  _Float16* sH = smem + 64 * 768;               // 16 KB
  __shared__ int sSrc[64], sDst[64];

  const int tid = threadIdx.x;
  const size_t e0 = (size_t)blockIdx.x * 64;

  if (tid < 64) {
    sSrc[tid] = edge_index[e0 + tid];
    sDst[tid] = edge_index[(size_t)NE + e0 + tid];
  }
  __syncthreads();

  // 64 rows x 48 sixteen-wide chunks = 3072 chunks; 12 per thread
#pragma unroll
  for (int i = 0; i < 12; ++i) {
    const int cid  = tid + 256 * i;       // 0..3071
    const int row  = cid / 48;
    const int col0 = (cid % 48) * 16;
    const int seg  = col0 >> 7;           // which 128-wide segment
    const int off  = col0 & 127;
    const _Float16* p;
    switch (seg) {
      case 0:  p = nf0 + (size_t)sSrc[row] * 128 + off; break;
      case 1:  p = nf  + (size_t)sSrc[row] * 128 + off; break;
      case 2:  p = nf0 + (size_t)sDst[row] * 128 + off; break;
      case 3:  p = nf  + (size_t)sDst[row] * 128 + off; break;
      case 4:  p = ef0 + (e0 + row) * 128 + off;        break;
      default: p = ef  + (e0 + row) * 128 + off;        break;
    }
    *(v16h*)(sA + (size_t)row * 768 + col0) = *(const v16h*)p;
  }
  __syncthreads();

  const int lane = tid & 31;
  const int col  = lane & 15;
  const int hi   = lane >> 4;
  const int n0   = (tid >> 5) * 16;

  // GEMM1: 64x768 @ 768x128 (4 M-tiles per wave)
  v8f c[4];
#pragma unroll
  for (int mt = 0; mt < 4; ++mt) c[mt] = (v8f){};
  for (int k0 = 0; k0 < 768; k0 += 32) {
    v16h b = load_b_tile(W1, k0, n0);
#pragma unroll
    for (int mt = 0; mt < 4; ++mt) {
      v16h a = load_a_tile(sA + (size_t)mt * 16 * 768 + k0, 768);
      c[mt] = wmma_step(a, b, c[mt]);
    }
  }
#pragma unroll
  for (int mt = 0; mt < 4; ++mt) {
#pragma unroll
    for (int r = 0; r < 8; ++r) {
      float v = c[mt][r] + b1[n0 + col];
      v = v > 0.f ? v : 0.f;
      sH[(size_t)(mt * 16 + r + 8 * hi) * 128 + n0 + col] = (_Float16)v;
    }
  }
  __syncthreads();

  // GEMM2: 64x128 @ 128x128
  v8f c2[4];
#pragma unroll
  for (int mt = 0; mt < 4; ++mt) c2[mt] = (v8f){};
  for (int k0 = 0; k0 < 128; k0 += 32) {
    v16h b = load_b_tile(W2, k0, n0);
#pragma unroll
    for (int mt = 0; mt < 4; ++mt) {
      v16h a = load_a_tile(sH + (size_t)mt * 16 * 128 + k0, 128);
      c2[mt] = wmma_step(a, b, c2[mt]);
    }
  }
#pragma unroll
  for (int mt = 0; mt < 4; ++mt) {
#pragma unroll
    for (int r = 0; r < 8; ++r) {
      const int row = mt * 16 + r + 8 * hi;
      float v = c2[mt][r] + b2[n0 + col];
      v = v > 0.f ? v : 0.f;
      ef_out[(e0 + row) * 128 + n0 + col] = (_Float16)v;
      unsafeAtomicAdd(&aggr[(size_t)sDst[row] * 128 + n0 + col], v);
    }
  }
}

// ---- node update: nf_new = relu([nf0|nf|aggr] @ Wn[384,128] + b) -----------
__global__ void mp_node_kernel(const _Float16* __restrict__ nf0,
                               const _Float16* __restrict__ nf,
                               const float* __restrict__ aggr,
                               const _Float16* __restrict__ W,
                               const float* __restrict__ bias,
                               _Float16* __restrict__ nf_out) {
  __shared__ _Float16 sA[16 * 384];   // 12 KB
  const int tid = threadIdx.x;
  const size_t m0 = (size_t)blockIdx.x * 16;

#pragma unroll
  for (int i = 0; i < 2; ++i) {
    const int cid = tid + 256 * i;    // 0..383 valid
    if (cid < 384) {
      const int row  = cid / 24;
      const int col0 = (cid % 24) * 16;
      const int seg  = col0 >> 7;
      const int off  = col0 & 127;
      const size_t node = m0 + row;
      if (seg < 2) {
        const _Float16* p = (seg == 0 ? nf0 : nf) + node * 128 + off;
        *(v16h*)(sA + (size_t)row * 384 + col0) = *(const v16h*)p;
      } else {
        const float* p = aggr + node * 128 + off;
#pragma unroll
        for (int j = 0; j < 16; ++j)
          sA[(size_t)row * 384 + col0 + j] = (_Float16)p[j];
      }
    }
  }
  __syncthreads();

  const int n0 = (tid >> 5) * 16;
  v8f c = {};
  for (int k0 = 0; k0 < 384; k0 += 32) {
    v16h a = load_a_tile(sA + k0, 384);
    v16h b = load_b_tile(W, k0, n0);
    c = wmma_step(a, b, c);
  }
  store_c_f16(c, bias, nf_out, m0, n0, /*relu=*/1);
}

// ---- small head kernels ----------------------------------------------------
__global__ void head_out1(const _Float16* __restrict__ h,
                          const float* __restrict__ w, const float* __restrict__ b,
                          const float* __restrict__ add,
                          float* __restrict__ out, int n) {
  const int i = blockIdx.x * blockDim.x + threadIdx.x;
  if (i >= n) return;
  float s = b[0];
  const _Float16* hp = h + (size_t)i * 128;
#pragma unroll 8
  for (int k = 0; k < 128; ++k) s += (float)hp[k] * w[k];
  out[i] = s + (add ? add[i] : 0.f);
}

__global__ void head_cls(const _Float16* __restrict__ h,
                         const float* __restrict__ W,   // [128,17]
                         const float* __restrict__ b,   // [17]
                         float* __restrict__ out, int n) {
  const int idx = blockIdx.x * blockDim.x + threadIdx.x;
  if (idx >= n * 17) return;
  const int i = idx / 17, c = idx % 17;
  float s = b[c];
  const _Float16* hp = h + (size_t)i * 128;
#pragma unroll 8
  for (int k = 0; k < 128; ++k) s += (float)hp[k] * W[k * 17 + c];
  out[idx] = s;
}

// ---- conversion / padding kernels -----------------------------------------
__global__ void cvt_f32_to_f16(const float* __restrict__ in,
                               _Float16* __restrict__ out, long n) {
  const long i = blockIdx.x * (long)blockDim.x + threadIdx.x;
  if (i < n) out[i] = (_Float16)in[i];
}

__global__ void pad_rows_cvt(const float* __restrict__ in,
                             _Float16* __restrict__ out,
                             int rin, int rout, int cols) {
  const int i = blockIdx.x * blockDim.x + threadIdx.x;
  if (i >= rout * cols) return;
  const int r = i / cols, c = i % cols;
  out[i] = (r < rin) ? (_Float16)in[r * cols + c] : (_Float16)0.f;
}

__global__ void pad_cols_cvt(const float* __restrict__ in,
                             _Float16* __restrict__ out,
                             long m, int cin, int cout) {
  const long i = blockIdx.x * (long)blockDim.x + threadIdx.x;
  if (i >= m * cout) return;
  const long r = i / cout; const int c = (int)(i % cout);
  out[i] = (c < cin) ? (_Float16)in[r * cin + c] : (_Float16)0.f;
}

// ---------------------------------------------------------------------------
extern "C" void kernel_launch(void* const* d_in, const int* in_sizes, int n_in,
                              void* d_out, int out_size, void* d_ws, size_t ws_size,
                              hipStream_t stream) {
  (void)in_sizes; (void)n_in; (void)out_size; (void)ws_size;

  // --- inputs (setup_inputs dict order; params flattened in insertion order,
  //     each MLP layer as (W, b)) ---
  const float* x          = (const float*)d_in[0];   // [NN,128]
  const float* edge_attr  = (const float*)d_in[1];   // [NE,17]
  const int*   edge_index = (const int*)  d_in[2];   // [2,NE]
  const float* joint_tags = (const float*)d_in[4];   // [NN]
  const float* ne_W1 = (const float*)d_in[5],  *ne_b1 = (const float*)d_in[6];
  const float* ne_W2 = (const float*)d_in[7],  *ne_b2 = (const float*)d_in[8];
  const float* ee_W1 = (const float*)d_in[9],  *ee_b1 = (const float*)d_in[10];
  const float* ee_W2 = (const float*)d_in[11], *ee_b2 = (const float*)d_in[12];
  const float* me_W1 = (const float*)d_in[13], *me_b1 = (const float*)d_in[14];
  const float* me_W2 = (const float*)d_in[15], *me_b2 = (const float*)d_in[16];
  const float* mn_W  = (const float*)d_in[17], *mn_b  = (const float*)d_in[18];
  const float* tg_W1 = (const float*)d_in[19], *tg_b1 = (const float*)d_in[20];
  const float* tg_W2 = (const float*)d_in[21], *tg_b2 = (const float*)d_in[22];
  const float* nc_W1 = (const float*)d_in[23], *nc_b1 = (const float*)d_in[24];
  const float* nc_W2 = (const float*)d_in[25], *nc_b2 = (const float*)d_in[26];
  const float* cl_W1 = (const float*)d_in[27], *cl_b1 = (const float*)d_in[28];
  const float* cl_W2 = (const float*)d_in[29], *cl_b2 = (const float*)d_in[30];

  float* out_node = (float*)d_out;              // [NN]
  float* out_cls  = out_node + NN;              // [NN,17]
  float* out_tag  = out_cls + (size_t)NN * 17;  // [NN]

  // --- workspace bump allocator (256B aligned) ---
  char* wsp = (char*)d_ws;
  auto alloc = [&](size_t bytes) -> void* {
    uintptr_t p = ((uintptr_t)wsp + 255) & ~(uintptr_t)255;
    wsp = (char*)(p + bytes);
    return (void*)p;
  };
  const size_t NFH = (size_t)NN * 128 * sizeof(_Float16);
  const size_t EFH = (size_t)NE * 128 * sizeof(_Float16);

  _Float16* x16  = (_Float16*)alloc(NFH);
  _Float16* ea16 = (_Float16*)alloc((size_t)NE * 32 * sizeof(_Float16));
  _Float16* nfA  = (_Float16*)alloc(NFH);   // nf0 (embedding, fixed)
  _Float16* nfB  = (_Float16*)alloc(NFH);
  _Float16* nfC  = (_Float16*)alloc(NFH);
  _Float16* ntmp = (_Float16*)alloc(NFH);   // MLP intermediates / head hidden
  _Float16* efA  = (_Float16*)alloc(EFH);   // ef0 (embedding, fixed)
  _Float16* efB  = (_Float16*)alloc(EFH);
  _Float16* efC  = (_Float16*)alloc(EFH);
  float*    aggr = (float*)   alloc((size_t)NN * 128 * sizeof(float));
  _Float16* wne1 = (_Float16*)alloc(128 * 128 * 2);
  _Float16* wne2 = (_Float16*)alloc(128 * 128 * 2);
  _Float16* wee1 = (_Float16*)alloc(32  * 128 * 2);  // padded 17->32 rows
  _Float16* wee2 = (_Float16*)alloc(128 * 128 * 2);
  _Float16* wme1 = (_Float16*)alloc(768 * 128 * 2);
  _Float16* wme2 = (_Float16*)alloc(128 * 128 * 2);
  _Float16* wmn  = (_Float16*)alloc(384 * 128 * 2);
  _Float16* wtg1 = (_Float16*)alloc(128 * 128 * 2);
  _Float16* wnc1 = (_Float16*)alloc(128 * 128 * 2);
  _Float16* wcl1 = (_Float16*)alloc(128 * 128 * 2);

  auto cvt = [&](const float* src, _Float16* dst, long n) {
    cvt_f32_to_f16<<<(unsigned)((n + 255) / 256), 256, 0, stream>>>(src, dst, n);
  };

  // --- fp32 -> f16 staging ---
  cvt(x, x16, (long)NN * 128);
  pad_cols_cvt<<<(unsigned)(((long)NE * 32 + 255) / 256), 256, 0, stream>>>(
      edge_attr, ea16, NE, 17, 32);
  cvt(ne_W1, wne1, 128 * 128);  cvt(ne_W2, wne2, 128 * 128);
  pad_rows_cvt<<<(32 * 128 + 255) / 256, 256, 0, stream>>>(ee_W1, wee1, 17, 32, 128);
  cvt(ee_W2, wee2, 128 * 128);
  cvt(me_W1, wme1, 768 * 128);  cvt(me_W2, wme2, 128 * 128);
  cvt(mn_W,  wmn,  384 * 128);
  cvt(tg_W1, wtg1, 128 * 128);  cvt(nc_W1, wnc1, 128 * 128);
  cvt(cl_W1, wcl1, 128 * 128);

  const int NG  = NN / 16;   // 625 node tiles (TM=1)
  const int EG4 = NE / 64;   // 2500 edge blocks (TM=4)
  auto gemm_n = [&](const _Float16* A, const _Float16* W, const float* b,
                    _Float16* out, int K, int relu) {
    gemm_tiled<1><<<NG, 256, (size_t)16 * K * sizeof(_Float16), stream>>>(
        A, W, b, out, K, relu);
  };
  auto gemm_e = [&](const _Float16* A, const _Float16* W, const float* b,
                    _Float16* out, int K, int relu) {
    gemm_tiled<4><<<EG4, 256, (size_t)64 * K * sizeof(_Float16), stream>>>(
        A, W, b, out, K, relu);
  };

  // --- embeddings (nf0 = nfA, ef0 = efA) ---
  gemm_n(x16,  wne1, ne_b1, ntmp, 128, 1);
  gemm_n(ntmp, wne2, ne_b2, nfA,  128, 1);
  gemm_e(ea16, wee1, ee_b1, efB,  32,  1);   // efB as scratch
  gemm_e(efB,  wee2, ee_b2, efA,  128, 1);

  // --- 4 message-passing layers (each an edge step + a node step) ---
  const size_t mp_lds = (size_t)(64 * 768 + 64 * 128) * sizeof(_Float16); // 112 KB
  _Float16* nf_cur = nfA;   // nf == nf0 initially
  _Float16* ef_cur = efA;   // ef == ef0 initially
  for (int layer = 0; layer < 4; ++layer) {
    _Float16* ef_next = (ef_cur == efB) ? efC : efB;
    _Float16* nf_next = (nf_cur == nfB) ? nfC : nfB;
    hipMemsetAsync(aggr, 0, (size_t)NN * 128 * sizeof(float), stream);
    mp_edge_kernel<<<NE / 64, 256, mp_lds, stream>>>(nfA, nf_cur, efA, ef_cur,
                                                     edge_index,
                                                     wme1, me_b1, wme2, me_b2,
                                                     ef_next, aggr);
    mp_node_kernel<<<NG, 256, 0, stream>>>(nfA, nf_cur, aggr, wmn, mn_b, nf_next);
    nf_cur = nf_next;
    ef_cur = ef_next;
    if (layer == 1) {   // tag head after the first two layers (with joint_tags skip)
      gemm_n(nf_cur, wtg1, tg_b1, ntmp, 128, 1);
      head_out1<<<(NN + 255) / 256, 256, 0, stream>>>(ntmp, tg_W2, tg_b2,
                                                      joint_tags, out_tag, NN);
    }
  }

  // --- node / class heads ---
  gemm_n(nf_cur, wnc1, nc_b1, ntmp, 128, 1);
  head_out1<<<(NN + 255) / 256, 256, 0, stream>>>(ntmp, nc_W2, nc_b2,
                                                  nullptr, out_node, NN);
  gemm_n(nf_cur, wcl1, cl_b1, ntmp, 128, 1);
  head_cls<<<(NN * 17 + 255) / 256, 256, 0, stream>>>(ntmp, cl_W2, cl_b2,
                                                      out_cls, NN);
}
